// PPNet_63788854280539
// MI455X (gfx1250) — matrix operand order repaired and checked by reading, not compiled, not run on hardware
//
#include <hip/hip_runtime.h>
#include <hip/hip_bf16.h>
#include <math.h>

// ---------------------------------------------------------------------------
// PPNet / pruned ViT-B16 forward for MI455X (gfx1250, wave32, WMMA f16)
// Guard-free WMMA GEMM, 2x2 tile blocking (32x32 C per wave).
// ---------------------------------------------------------------------------

typedef __attribute__((ext_vector_type(16))) _Float16 v16h;
typedef __attribute__((ext_vector_type(8)))  _Float16 v8h;
typedef __attribute__((ext_vector_type(8)))  float    v8f;

static constexpr int kB  = 64;
static constexpr int kD  = 768;
static constexpr int kH  = 12;
static constexpr int kL  = 12;
static constexpr int kNP = 196;
static constexpr int kDH = 64;
static constexpr int kNp0 = 224;   // pad(197) to multiple of 32
static constexpr int kMp  = 224;   // pad(196) to multiple of 32 (patch GEMM)

// ---------------------------------------------------------------------------
// Batched WMMA GEMM:  C = act( scale * (A @ B^T) + bias + resid )
//   A  : f16, M x K row-major, lda    (M%32==0, N%32==0, K%32==0)
//   Bt : f16, N x K row-major, ldb    ("transposed" storage, contiguous K)
//   C  : f32 or f16 (exactly one non-null), ldc
// batch offset = (z/nb2)*s?1 + (z%nb2)*s?2.
// One wave computes a 32x32 C macro-tile (2x2 WMMA tiles): A and B fragments
// are each reused twice -> 4x b128-pair loads per 4 WMMAs.
// A/B use contiguous K per lane half (k = 16*h + s), C/D per ISA 7.12.2; no bounds checks (padded shapes).
// ---------------------------------------------------------------------------
__global__ __launch_bounds__(32) void k_wmma_gemm(
    const _Float16* __restrict__ A, int lda, long sA1, long sA2,
    const _Float16* __restrict__ Bt, int ldb, long sB1, long sB2,
    float* __restrict__ Cf32, _Float16* __restrict__ Cf16,
    int ldc, long sC1, long sC2,
    const float* __restrict__ bias, const float* __restrict__ resid,
    int K, int nb2, float scale, int act)
{
    const int lane = threadIdx.x;                  // wave32
    const int tn0  = blockIdx.x * 32;              // N tiles: tn0, tn0+16
    const int tm0  = blockIdx.y * 32;              // M tiles: tm0, tm0+16
    const int z    = blockIdx.z;

    const long offA = (long)(z / nb2) * sA1 + (long)(z % nb2) * sA2;
    const long offB = (long)(z / nb2) * sB1 + (long)(z % nb2) * sB2;
    const long offC = (long)(z / nb2) * sC1 + (long)(z % nb2) * sC2;

    const int ln    = lane & 15;
    const int aKsel = (lane >> 4) * 16;            // A lane-half K offset
    const int bKsel = (lane >> 4) * 16;            // B lane-half K offset

    const _Float16* __restrict__ Ap0 = A  + offA + (long)(tm0 + ln) * lda + aKsel;
    const _Float16* __restrict__ Ap1 = Ap0 + (long)16 * lda;
    const _Float16* __restrict__ Bp0 = Bt + offB + (long)(tn0 + ln) * ldb + bKsel;
    const _Float16* __restrict__ Bp1 = Bp0 + (long)16 * ldb;

    v8f c00 = {0.f,0.f,0.f,0.f,0.f,0.f,0.f,0.f};
    v8f c01 = c00, c10 = c00, c11 = c00;

    for (int k0 = 0; k0 < K; k0 += 32) {
        v8h  a0l = *(const v8h*)(Ap0 + k0);        // K +0..7   (this lane half)
        v8h  a0h = *(const v8h*)(Ap0 + k0 + 8);    // K +8..15  (this lane half)
        v8h  a1l = *(const v8h*)(Ap1 + k0);
        v8h  a1h = *(const v8h*)(Ap1 + k0 + 8);
        v16h a0  = __builtin_shufflevector(a0l, a0h,
                     0,1,2,3,4,5,6,7,8,9,10,11,12,13,14,15);
        v16h a1  = __builtin_shufflevector(a1l, a1h,
                     0,1,2,3,4,5,6,7,8,9,10,11,12,13,14,15);
        v16h b0  = *(const v16h*)(Bp0 + k0);       // 16 contiguous K per lane
        v16h b1  = *(const v16h*)(Bp1 + k0);
        c00 = __builtin_amdgcn_wmma_f32_16x16x32_f16(
                  false, a0, false, b0, (short)0, c00, false, false);
        c01 = __builtin_amdgcn_wmma_f32_16x16x32_f16(
                  false, a0, false, b1, (short)0, c01, false, false);
        c10 = __builtin_amdgcn_wmma_f32_16x16x32_f16(
                  false, a1, false, b0, (short)0, c10, false, false);
        c11 = __builtin_amdgcn_wmma_f32_16x16x32_f16(
                  false, a1, false, b1, (short)0, c11, false, false);
    }

    const int mhalf = (lane >> 4) * 8;             // C/D row sub-base (ISA layout)
#pragma unroll
    for (int mt = 0; mt < 2; ++mt) {
#pragma unroll
        for (int nt = 0; nt < 2; ++nt) {
            const v8f& c = mt ? (nt ? c11 : c10) : (nt ? c01 : c00);
            const int gn = tn0 + 16 * nt + ln;
            const int mb = tm0 + 16 * mt + mhalf;
#pragma unroll
            for (int r = 0; r < 8; ++r) {
                const int gm = mb + r;
                long idx = offC + (long)gm * ldc + gn;
                float v = c[r] * scale;
                if (bias)  v += bias[gn];
                if (resid) v += resid[idx];
                if (act == 1) {                    // tanh-approx GELU
                    float xx = v;
                    v = 0.5f * xx * (1.0f + tanhf(0.7978845608f *
                            (xx + 0.044715f * xx * xx * xx)));
                } else if (act == 2) {             // sigmoid
                    v = 1.0f / (1.0f + expf(-v));
                }
                if (Cf32) Cf32[idx] = v;
                else      Cf16[idx] = (_Float16)v;
            }
        }
    }
}

// ---------------------------------------------------------------------------
// conversions / transposes
// ---------------------------------------------------------------------------
__global__ void k_cvt_f16(const float* __restrict__ s, _Float16* __restrict__ d, long n)
{
    for (long i = (long)blockIdx.x * blockDim.x + threadIdx.x;
         i < n; i += (long)gridDim.x * blockDim.x)
        d[i] = (_Float16)s[i];
}

// src: (L, K, N) f32 -> dst: (L, N, K) f16
__global__ void k_cvt_T(const float* __restrict__ s, _Float16* __restrict__ d,
                        int K, int N, long total)
{
    for (long t = (long)blockIdx.x * blockDim.x + threadIdx.x;
         t < total; t += (long)gridDim.x * blockDim.x) {
        int  kidx = (int)(t % K);
        long r    = t / K;
        int  n    = (int)(r % N);
        long l    = r / N;
        d[t] = (_Float16)s[(l * K + kidx) * (long)N + n];
    }
}

// V transpose: vT[(b*H+h)*64*Np + d*Np + kk] = qkv[(b*Np+kk)*2304 + 1536 + h*64 + d]
__global__ void k_vtrans(const _Float16* __restrict__ qkv,
                         _Float16* __restrict__ vT, int Np, long total)
{
    for (long t = (long)blockIdx.x * blockDim.x + threadIdx.x;
         t < total; t += (long)gridDim.x * blockDim.x) {
        int  kk = (int)(t % Np);
        long r  = t / Np;
        int  d  = (int)(r % kDH);
        long r2 = r / kDH;
        int  h  = (int)(r2 % kH);
        int  b  = (int)(r2 / kH);
        vT[t] = qkv[((long)b * Np + kk) * (3 * kD) + 2 * kD + h * kDH + d];
    }
}

// im2col (f16) with per-batch row stride kMp (=224): rows 196..223 stay stale
__global__ void k_im2col(const float* __restrict__ x, _Float16* __restrict__ o)
{
    long t = (long)blockIdx.x * blockDim.x + threadIdx.x;
    const long total = (long)kB * kNP * kD;
    if (t >= total) return;
    int col = (int)(t % kD);
    long row = t / kD;
    int b = (int)(row / kNP), pp = (int)(row % kNP);
    int pi = pp / 14, pj = pp % 14;
    int cch = col / 256, r = col % 256, ph = r / 16, pw = r % 16;
    o[((long)b * kMp + pp) * kD + col] =
        (_Float16)x[(((long)b * 3 + cch) * 224 + (pi * 16 + ph)) * 224 +
                    (pj * 16 + pw)];
}

// tok[b,0,:] = cls + pos[0,:]; tok[b,1<=n<197,:] += pos[n,:]   (stride kNp0)
// NOTE: must run AFTER patch GEMM (its padded rows touch row 0 of image b+1;
// this kernel re-assigns every row 0).
__global__ void k_add_pos_cls(float* __restrict__ tok,
                              const float* __restrict__ cls,
                              const float* __restrict__ pos)
{
    long t = (long)blockIdx.x * blockDim.x + threadIdx.x;
    const long total = (long)kB * 197 * kD;
    if (t >= total) return;
    int d = (int)(t % kD);
    long r = t / kD;
    int n = (int)(r % 197);
    int b = (int)(r / 197);
    long idx = ((long)b * kNp0 + n) * kD + d;
    if (n == 0) tok[idx] = cls[d] + pos[d];
    else        tok[idx] += pos[(long)n * kD + d];
}

// LayerNorm (biased var, eps 1e-6) with affine -> f16 (runs on padding rows too)
__global__ __launch_bounds__(256) void k_ln(const float* __restrict__ tok,
                                            const float* __restrict__ g,
                                            const float* __restrict__ b,
                                            _Float16* __restrict__ o)
{
    __shared__ float s1[256], s2[256];
    long row = blockIdx.x;
    const float* xr = tok + row * kD;
    int t = threadIdx.x;
    float a0 = 0.f, a1 = 0.f;
    for (int d = t; d < kD; d += 256) { float v = xr[d]; a0 += v; a1 += v * v; }
    s1[t] = a0; s2[t] = a1; __syncthreads();
    for (int s = 128; s > 0; s >>= 1) {
        if (t < s) { s1[t] += s1[t + s]; s2[t] += s2[t + s]; }
        __syncthreads();
    }
    float mean = s1[0] * (1.0f / kD);
    float var  = s2[0] * (1.0f / kD) - mean * mean;
    float inv  = rsqrtf(fmaxf(var, 0.f) + 1e-6f);
    for (int d = t; d < kD; d += 256)
        o[row * kD + d] = (_Float16)((xr[d] - mean) * inv * g[d] + b[d]);
}

// softmax over first Ncur cols -> f16 attn; cols [Ncur, Np) zeroed
__global__ __launch_bounds__(256) void k_softmax(const float* __restrict__ sc,
                                                 _Float16* __restrict__ at,
                                                 int Ncur, int Np)
{
    __shared__ float red[256];
    long base = (long)blockIdx.x * Np;
    int t = threadIdx.x;
    float mx = -1e30f;
    for (int i = t; i < Ncur; i += 256) mx = fmaxf(mx, sc[base + i]);
    red[t] = mx; __syncthreads();
    for (int s = 128; s > 0; s >>= 1) {
        if (t < s) red[t] = fmaxf(red[t], red[t + s]);
        __syncthreads();
    }
    mx = red[0]; __syncthreads();
    float sum = 0.f;
    for (int i = t; i < Ncur; i += 256) sum += expf(sc[base + i] - mx);
    red[t] = sum; __syncthreads();
    for (int s = 128; s > 0; s >>= 1) {
        if (t < s) red[t] += red[t + s];
        __syncthreads();
    }
    float inv = 1.0f / red[0];
    for (int i = t; i < Np; i += 256)
        at[base + i] = (i < Ncur)
            ? (_Float16)(expf(sc[base + i] - mx) * inv) : (_Float16)0.0f;
}

// cls_attn *= mean_h attn[b,h,0,1:]; then renormalize per image
__global__ __launch_bounds__(256) void k_cls_update(const _Float16* __restrict__ at,
                                                    float* __restrict__ cls,
                                                    int Ncur, int Np)
{
    __shared__ float red[256];
    int b = blockIdx.x, t = threadIdx.x;
    int n = Ncur - 1;
    const long NpNp = (long)Np * Np;
    float part = 0.f;
    for (int i = t; i < n; i += 256) {
        float s = 0.f;
        for (int h = 0; h < kH; ++h)
            s += (float)at[(long)(b * kH + h) * NpNp + 1 + i];
        float v = cls[(long)b * kNP + i] * (s * (1.0f / kH));
        cls[(long)b * kNP + i] = v;
        part += v;
    }
    red[t] = part; __syncthreads();
    for (int s = 128; s > 0; s >>= 1) {
        if (t < s) red[t] += red[t + s];
        __syncthreads();
    }
    float inv = 1.0f / red[0];
    for (int i = t; i < n; i += 256) cls[(long)b * kNP + i] *= inv;
}

__global__ void k_cls_init(float* cls)
{
    long t = (long)blockIdx.x * blockDim.x + threadIdx.x;
    if (t < (long)kB * kNP) cls[t] = 1.0f;
}

// stable top-k (matches lax.top_k ties) + sorted-index compaction
__global__ __launch_bounds__(256) void k_topk(const float* __restrict__ cls,
                                              float* __restrict__ cls2,
                                              int* __restrict__ map, int n, int kk)
{
    __shared__ float sv[256];
    __shared__ int   sk[256];
    int b = blockIdx.x, t = threadIdx.x;
    sv[t] = (t < n) ? cls[(long)b * kNP + t] : -1e30f;
    __syncthreads();
    int keep = 0;
    if (t < n) {
        int rank = 0;
        float vt = sv[t];
        for (int j = 0; j < n; ++j) {
            float vj = sv[j];
            if (vj > vt || (vj == vt && j < t)) ++rank;
        }
        keep = (rank < kk);
    }
    sk[t] = keep; __syncthreads();
    if (keep) {
        int pos = 0;
        for (int j = 0; j < t; ++j) pos += sk[j];
        map[(long)b * kNP + pos] = t;
        cls2[(long)b * kNP + pos] = sv[t];
    }
}

// gather kept token rows (row 0 = CLS always kept); strides NpOld -> NpNew
__global__ __launch_bounds__(256) void k_gather(const float* __restrict__ ti,
                                                float* __restrict__ to,
                                                const int* __restrict__ map,
                                                int NpOld, int NpNew)
{
    int b = blockIdx.x, r = blockIdx.y;               // r in [0, kk]
    int src = (r == 0) ? 0 : 1 + map[(long)b * kNP + (r - 1)];
    const float* s = ti + ((long)b * NpOld + src) * kD;
    float* d = to + ((long)b * NpNew + r) * kD;
    for (int i = threadIdx.x; i < kD; i += 256) d[i] = s[i];
}

// tok rows 1..121 (stride Np) -> f16 matrix (B*121, 768)
__global__ __launch_bounds__(256) void k_img_rows(const float* __restrict__ tok,
                                                  _Float16* __restrict__ o, int Np)
{
    int b = blockIdx.x, r = blockIdx.y;               // r in [0,121)
    const float* s = tok + ((long)b * Np + 1 + r) * kD;
    _Float16* d = o + ((long)b * 121 + r) * kD;
    for (int i = threadIdx.x; i < kD; i += 256) d[i] = (_Float16)s[i];
}

// prototype L2 distances + log activation + spatial max -> act[b, 20]
__global__ __launch_bounds__(128) void k_proto(const float* __restrict__ imgf,
                                               const float* __restrict__ pw,
                                               float* __restrict__ act)
{
    __shared__ float sh[128], red[128], p2s[20];
    int b = blockIdx.x, t = threadIdx.x;
    if (t < 20) {
        float s = 0.f;
        for (int c = 0; c < 128; ++c) { float w = pw[t * 128 + c]; s += w * w; }
        p2s[t] = s;
    }
    float best = -1e30f;
    for (int hw = 0; hw < 121; ++hw) {
        __syncthreads();
        float v = imgf[((long)b * 121 + hw) * 128 + t];
        sh[t] = v; red[t] = v * v;
        __syncthreads();
        for (int s = 64; s > 0; s >>= 1) {
            if (t < s) red[t] += red[t + s];
            __syncthreads();
        }
        if (t < 20) {
            float xp = 0.f;
            for (int c = 0; c < 128; ++c) xp += sh[c] * pw[t * 128 + c];
            float dist = fmaxf(red[0] - 2.0f * xp + p2s[t], 0.0f);
            float a = logf((dist + 1.0f) / (dist + 1e-4f));
            best = fmaxf(best, a);
        }
    }
    if (t < 20) act[(long)b * 20 + t] = best;
}

// out[b, j] = sum_p act[b,p] * last_w[j,p]   (last_w is 2x20)
__global__ void k_final(const float* __restrict__ act,
                        const float* __restrict__ lw, float* __restrict__ out)
{
    int t = (int)(blockIdx.x * blockDim.x + threadIdx.x);
    if (t >= kB * 2) return;
    int b = t / 2, j = t % 2;
    float s = 0.f;
    for (int pp = 0; pp < 20; ++pp) s += act[b * 20 + pp] * lw[j * 20 + pp];
    out[t] = s;
}

// ---------------------------------------------------------------------------
// Host orchestration
// ---------------------------------------------------------------------------
static inline void run_gemm(hipStream_t st,
    const _Float16* A, int lda, long sA1, long sA2,
    const _Float16* Bt, int ldb, long sB1, long sB2,
    float* Cf32, _Float16* Cf16, int ldc, long sC1, long sC2,
    const float* bias, const float* resid,
    int M, int N, int K, int batches, int nb2, float scale, int act)
{
    dim3 g(N / 32, M / 32, batches);   // caller guarantees M%32==0, N%32==0
    k_wmma_gemm<<<g, 32, 0, st>>>(A, lda, sA1, sA2, Bt, ldb, sB1, sB2,
                                  Cf32, Cf16, ldc, sC1, sC2, bias, resid,
                                  K, nb2, scale, act);
}

extern "C" void kernel_launch(void* const* d_in, const int* in_sizes, int n_in,
                              void* d_out, int out_size, void* d_ws, size_t ws_size,
                              hipStream_t stream)
{
    const float* x       = (const float*)d_in[0];
    const float* patch_w = (const float*)d_in[1];
    const float* patch_b = (const float*)d_in[2];
    const float* cls_tok = (const float*)d_in[3];
    const float* pos     = (const float*)d_in[4];
    const float* ln1_g   = (const float*)d_in[5];
    const float* ln1_b   = (const float*)d_in[6];
    const float* Wqkv    = (const float*)d_in[7];
    const float* bqkv    = (const float*)d_in[8];
    const float* Wo      = (const float*)d_in[9];
    const float* bo      = (const float*)d_in[10];
    const float* ln2_g   = (const float*)d_in[11];
    const float* ln2_b   = (const float*)d_in[12];
    const float* W1      = (const float*)d_in[13];
    const float* b1      = (const float*)d_in[14];
    const float* W2      = (const float*)d_in[15];
    const float* b2      = (const float*)d_in[16];
    const float* addon_w = (const float*)d_in[17];
    const float* addon_b = (const float*)d_in[18];
    const float* protos  = (const float*)d_in[19];
    const float* last_w  = (const float*)d_in[20];
    float* out = (float*)d_out;

    // --- workspace carve (bump allocator, 256B aligned) ---
    char* wp = (char*)d_ws;
    auto alloc = [&](size_t bytes) -> void* {
        void* r = (void*)wp;
        wp += (bytes + 255) & ~(size_t)255;
        return r;
    };
    const long tokElems = (long)kB * kNp0 * kD;
    float*    tokA   = (float*)   alloc(tokElems * 4);
    float*    tokB   = (float*)   alloc(tokElems * 4);
    _Float16* h16    = (_Float16*)alloc(tokElems * 2);
    _Float16* qkv16  = (_Float16*)alloc((long)kB * kNp0 * 3 * kD * 2);
    float*    scores = (float*)   alloc((long)kB * kH * kNp0 * kNp0 * 4);
    _Float16* attn16 = (_Float16*)alloc((long)kB * kH * kNp0 * kNp0 * 2);
    _Float16* vT16   = (_Float16*)alloc((long)kB * kH * kDH * kNp0 * 2);
    _Float16* o16    = (_Float16*)alloc(tokElems * 2);
    _Float16* m16    = (_Float16*)alloc((long)kB * kNp0 * 4 * kD * 2);
    _Float16* col16  = (_Float16*)alloc((long)kB * kMp * kD * 2);
    _Float16* wq16   = (_Float16*)alloc((long)kL * kD * 3 * kD * 2);   // (L, 2304, 768)
    _Float16* wo16   = (_Float16*)alloc((long)kL * kD * kD * 2);       // (L, 768, 768) T
    _Float16* w116   = (_Float16*)alloc((long)kL * kD * 4 * kD * 2);   // (L, 3072, 768)
    _Float16* w216   = (_Float16*)alloc((long)kL * 4 * kD * kD * 2);   // (L, 768, 3072)
    _Float16* pw16   = (_Float16*)alloc((long)kD * kD * 2);            // already N x K
    _Float16* ad16   = (_Float16*)alloc((long)128 * kD * 2);           // already N x K
    float*    clsA   = (float*)   alloc((long)kB * kNP * 4);
    float*    clsB   = (float*)   alloc((long)kB * kNP * 4);
    int*      map    = (int*)     alloc((long)kB * kNP * 4);
    _Float16* img16  = (_Float16*)alloc((long)kB * 121 * kD * 2);
    float*    imgf   = (float*)   alloc((long)kB * 121 * 128 * 4);
    float*    actv   = (float*)   alloc((long)kB * 20 * 4);

    auto gridFor = [](long n) -> int {
        long b = (n + 255) / 256;
        return (int)(b > 65535 ? 65535 : b);
    };
    auto cvt = [&](const float* s, _Float16* d, long n) {
        k_cvt_f16<<<gridFor(n), 256, 0, stream>>>(s, d, n);
    };
    auto cvtT = [&](const float* s, _Float16* d, int K, int N, int L) {
        long n = (long)L * K * N;
        k_cvt_T<<<gridFor(n), 256, 0, stream>>>(s, d, K, N, n);
    };
    // weights -> f16, stored N x K ("transposed") for contiguous-K WMMA loads
    cvtT(Wqkv, wq16, kD, 3 * kD, kL);
    cvtT(Wo,   wo16, kD, kD,     kL);
    cvtT(W1,   w116, kD, 4 * kD, kL);
    cvtT(W2,   w216, 4 * kD, kD, kL);
    cvt(patch_w, pw16, (long)kD * kD);     // (768 out, 768 in) is already N x K
    cvt(addon_w, ad16, (long)128 * kD);    // (128, 768) is already N x K

    // --- patch embedding: im2col + GEMM -> tok rows 1..196 (+pad), cls/pos ---
    {
        long total = (long)kB * kNP * kD;
        k_im2col<<<(int)((total + 255) / 256), 256, 0, stream>>>(x, col16);
        run_gemm(stream, col16, kD, (long)kMp * kD, 0,
                 pw16, kD, 0, 0,
                 tokA + kD, nullptr, kD, (long)kNp0 * kD, 0,
                 patch_b, nullptr, kMp, kD, kD, kB, 1, 1.0f, 0);
        total = (long)kB * 197 * kD;
        k_add_pos_cls<<<(int)((total + 255) / 256), 256, 0, stream>>>(tokA, cls_tok, pos);
        k_cls_init<<<(kB * kNP + 255) / 256, 256, 0, stream>>>(clsA);
    }

    float* tok = tokA; float* tokAlt = tokB;
    float* cls = clsA; float* clsAlt = clsB;
    int Ncur = 197, Np = kNp0;

    for (int i = 0; i < kL; ++i) {
        const int  rows = kB * Np;               // multiple of 32
        const long NpNp = (long)Np * Np;

        // h = LN1(tok) -> f16
        k_ln<<<rows, 256, 0, stream>>>(tok, ln1_g + i * kD, ln1_b + i * kD, h16);

        // qkv = h @ Wqkv + bqkv  -> f16 (rows x 2304)
        run_gemm(stream, h16, kD, 0, 0,
                 wq16 + (long)i * kD * 3 * kD, kD, 0, 0,
                 nullptr, qkv16, 3 * kD, 0, 0,
                 bqkv + i * 3 * kD, nullptr, rows, 3 * kD, kD, 1, 1, 1.0f, 0);

        // scores[b,h] = q @ k^T * DH^-0.5   (k rows are contiguous-K)
        run_gemm(stream, qkv16, 3 * kD, (long)Np * 3 * kD, kDH,
                 qkv16 + kD, 3 * kD, (long)Np * 3 * kD, kDH,
                 scores, nullptr, Np, (long)kH * NpNp, NpNp,
                 nullptr, nullptr, Np, Np, kDH, kB * kH, kH, 0.125f, 0);

        // softmax (valid cols) -> f16 attn, zero pad cols; cls bookkeeping
        k_softmax<<<kB * kH * Np, 256, 0, stream>>>(scores, attn16, Ncur, Np);
        k_cls_update<<<kB, 256, 0, stream>>>(attn16, cls, Ncur, Np);

        // vT[b,h,d,kk] for contiguous-K attn@v
        {
            long total = (long)kB * kH * kDH * Np;
            k_vtrans<<<gridFor(total), 256, 0, stream>>>(qkv16, vT16, Np, total);
        }
        // o[b, :, h*64..] = attn @ v -> f16 (rows x 768)
        run_gemm(stream, attn16, Np, (long)kH * NpNp, NpNp,
                 vT16, Np, (long)kH * kDH * Np, (long)kDH * Np,
                 nullptr, o16, kD, (long)Np * kD, kDH,
                 nullptr, nullptr, Np, kDH, Np, kB * kH, kH, 1.0f, 0);

        // tok += o @ Wo + bo
        run_gemm(stream, o16, kD, 0, 0,
                 wo16 + (long)i * kD * kD, kD, 0, 0,
                 tok, nullptr, kD, 0, 0,
                 bo + i * kD, tok, rows, kD, kD, 1, 1, 1.0f, 0);

        // h2 = LN2(tok) -> f16 ; MLP
        k_ln<<<rows, 256, 0, stream>>>(tok, ln2_g + i * kD, ln2_b + i * kD, h16);
        run_gemm(stream, h16, kD, 0, 0,
                 w116 + (long)i * kD * 4 * kD, kD, 0, 0,
                 nullptr, m16, 4 * kD, 0, 0,
                 b1 + i * 4 * kD, nullptr, rows, 4 * kD, kD, 1, 1, 1.0f, /*gelu*/1);
        run_gemm(stream, m16, 4 * kD, 0, 0,
                 w216 + (long)i * 4 * kD * kD, 4 * kD, 0, 0,
                 tok, nullptr, kD, 0, 0,
                 b2 + i * kD, tok, rows, kD, 4 * kD, 1, 1, 1.0f, 0);

        // token pruning after layers 9 and 11 (i == 8, 10)
        int kk = (i == 8) ? 144 : (i == 10) ? 121 : 0;
        if (kk) {
            int NpNew = (kk + 1 + 31) & ~31;     // 160, 128
            k_topk<<<kB, 256, 0, stream>>>(cls, clsAlt, map, Ncur - 1, kk);
            dim3 gg(kB, kk + 1);
            k_gather<<<gg, 256, 0, stream>>>(tok, tokAlt, map, Np, NpNew);
            { float* t0 = tok; tok = tokAlt; tokAlt = t0; }
            { float* c0 = cls; cls = clsAlt; clsAlt = c0; }
            Ncur = kk + 1; Np = NpNew;
        }
    }

    // Final top_k(cls_attn, 121) over exactly 121 tokens is the identity
    // permutation after index sort -> img = tok[:, 1:122, :] directly.
    {
        dim3 gg(kB, 121);
        k_img_rows<<<gg, 256, 0, stream>>>(tok, img16, Np);
        // img_f = sigmoid(img @ addon_w^T + addon_b)  (7744 x 128; 7744 = 32*242)
        run_gemm(stream, img16, kD, 0, 0,
                 ad16, kD, 0, 0,
                 imgf, nullptr, 128, 0, 0,
                 addon_b, nullptr, kB * 121, 128, kD, 1, 1, 1.0f, /*sigmoid*/2);
        k_proto<<<kB, 128, 0, stream>>>(imgf, protos, actv);
        k_final<<<1, 128, 0, stream>>>(actv, last_w, out);
    }
}
